// DifferentiableEpisodicMemory_72670846648627
// MI455X (gfx1250) — matrix operand order repaired and checked by reading, not compiled, not run on hardware
//
#include <hip/hip_runtime.h>
#include <stdint.h>

// ---------------------------------------------------------------------------
// CDNA5 / gfx1250 fused "episodic memory" block.
//   Q = x@Wq+bq ; Qn=l2norm(Q) ; s = Qn@KnT/sqrt(d) ; w=softmax(s)
//   R = w@V ; g = sigmoid([x,R]@Wg+bg) ; out = g*x + (1-g)*R
// All big GEMMs run on v_wmma_f32_16x16x32_bf16 (f32 accumulate).
// ---------------------------------------------------------------------------

typedef __attribute__((ext_vector_type(16))) __bf16 v16bf;
typedef __attribute__((ext_vector_type(8)))  float  v8f;

#define DIMD  1024
#define NSLOT 512
#define ROWS  32     // token rows per block

// ---- bf16 helpers (round-to-nearest-even via bit trick) -------------------
__device__ __forceinline__ uint16_t f2bf(float x) {
    uint32_t u = __float_as_uint(x);
    u += 0x7fffu + ((u >> 16) & 1u);
    return (uint16_t)(u >> 16);
}
__device__ __forceinline__ float bf2f(uint16_t b) {
    return __uint_as_float(((uint32_t)b) << 16);
}

// ---- WMMA fragment loaders ------------------------------------------------
// A (16x32 bf16): lane L holds row (L&15); h=L>>4 selects k half;
// VGPR v holds k = kb+16h+2v{,+1} => two 16B loads.
struct Frag { union { v16bf v; uint4 q[2]; }; };

__device__ __forceinline__ v16bf load_frag_a(const uint16_t* p) {
    Frag f;
    f.q[0] = *(const uint4*)(p);        // k = kb+16h .. +7
    f.q[1] = *(const uint4*)(p + 8);    // k = kb+16h+8 .. +15
    return f.v;
}
// B (32x16 bf16): lane n = L&15, h = L>>4 selects k half; VGPR v holds
// k = kb+16h+2v{,+1} => 32 contiguous bytes = two 16B loads.
__device__ __forceinline__ v16bf load_frag_b(const uint16_t* p) {
    Frag f;
    f.q[0] = *(const uint4*)(p);        // k = kb+16h .. +7
    f.q[1] = *(const uint4*)(p + 8);    // k = kb+16h+8 .. +15
    return f.v;
}

__device__ __forceinline__ v8f wmma_bf16(v16bf a, v16bf b, v8f c) {
    return __builtin_amdgcn_wmma_f32_16x16x32_bf16(
        /*neg_a=*/false, a, /*neg_b=*/false, b,
        /*c_mod=*/(short)0, c, /*reuse_a=*/false, /*reuse_b=*/false);
}

__device__ __forceinline__ v8f zero8() {
    v8f z = {0.f,0.f,0.f,0.f,0.f,0.f,0.f,0.f};
    return z;
}

// ---------------------------------------------------------------------------
// Prep kernel 1: transpose+convert f32 weight W[K][N] -> bf16 WT[N][K]
// ---------------------------------------------------------------------------
__global__ void convert_wT_kernel(const float* __restrict__ W,
                                  uint16_t* __restrict__ WT,
                                  int K, int N) {
    int idx = blockIdx.x * 256 + threadIdx.x;
    if (idx < K * N) {
        int n = idx / K;
        int k = idx - n * K;
        WT[idx] = f2bf(W[(size_t)k * N + n]);
    }
}

// ---------------------------------------------------------------------------
// Prep kernel 2: K = mem@Wk+bk (row-l2-normalized, bf16 Kn[slot][d])
//                V = mem@Wv+bv (bf16, transposed Vt[col][slot])
// Small GEMMs (<1% of FLOPs) -> scalar f32.
// ---------------------------------------------------------------------------
__global__ __launch_bounds__(256) void kv_kernel(
    const float* __restrict__ memv,
    const float* __restrict__ Wk, const float* __restrict__ bk,
    const float* __restrict__ Wv, const float* __restrict__ bv,
    uint16_t* __restrict__ Kn, uint16_t* __restrict__ Vt) {
    __shared__ float mrow[DIMD];
    __shared__ float red[256];
    const int slot = blockIdx.x;
    const int tid  = threadIdx.x;

    for (int i = tid; i < DIMD; i += 256)
        mrow[i] = memv[(size_t)slot * DIMD + i];
    __syncthreads();

    float ka[4], va[4];
#pragma unroll
    for (int j = 0; j < 4; ++j) {
        int c = tid + 256 * j;
        ka[j] = bk[c];
        va[j] = bv[c];
    }
    for (int kk = 0; kk < DIMD; ++kk) {
        float m = mrow[kk];
#pragma unroll
        for (int j = 0; j < 4; ++j) {
            int c = tid + 256 * j;
            ka[j] = fmaf(m, Wk[(size_t)kk * DIMD + c], ka[j]);
            va[j] = fmaf(m, Wv[(size_t)kk * DIMD + c], va[j]);
        }
    }
    float ss = 0.f;
#pragma unroll
    for (int j = 0; j < 4; ++j) ss += ka[j] * ka[j];
    red[tid] = ss;
    __syncthreads();
    for (int s = 128; s > 0; s >>= 1) {
        if (tid < s) red[tid] += red[tid + s];
        __syncthreads();
    }
    float scale = 1.0f / fmaxf(sqrtf(red[0]), 1e-12f);
#pragma unroll
    for (int j = 0; j < 4; ++j) {
        int c = tid + 256 * j;
        Kn[(size_t)slot * DIMD + c] = f2bf(ka[j] * scale);
        Vt[(size_t)c * NSLOT + slot] = f2bf(va[j]);
    }
}

// ---------------------------------------------------------------------------
// Main fused kernel: 256 threads = 8 waves. wave -> (mt = w>>2, nq = w&3).
// Each wave owns a 16-row M tile; 4 waves split the N dimension.
// LDS: xb(64K bf16) | qb(64K bf16, reused for R) | wf(64K f32) | wb(32K bf16)
// ---------------------------------------------------------------------------
__global__ __launch_bounds__(256) void fused_kernel(
    const float* __restrict__ x,      // [32768][1024] f32
    const float* __restrict__ bq,     // [1024]
    const float* __restrict__ bg,     // [1024]
    const uint16_t* __restrict__ WqT, // [1024][1024] bf16 (n-major)
    const uint16_t* __restrict__ WgT, // [1024][2048] bf16 (n-major)
    const uint16_t* __restrict__ Kn,  // [512][1024]  bf16
    const uint16_t* __restrict__ Vt,  // [1024][512]  bf16
    float* __restrict__ out) {
    extern __shared__ __align__(16) char smem_raw[];
    uint16_t* xb = (uint16_t*)smem_raw;            // ROWS*DIMD bf16
    uint16_t* qb = xb + ROWS * DIMD;               // ROWS*DIMD bf16 (Q, then R)
    float*    wf = (float*)(qb + ROWS * DIMD);     // ROWS*NSLOT f32 scores
    uint16_t* wb = (uint16_t*)(wf + ROWS * NSLOT); // ROWS*NSLOT bf16 softmax
    float* qscale = (float*)(wb + ROWS * NSLOT);   // ROWS
    float* rmax   = qscale + ROWS;                 // ROWS
    float* rsum   = rmax + ROWS;                   // ROWS

    const int tid  = threadIdx.x;
    const int wave = tid >> 5;
    const int lane = tid & 31;
    const int h    = lane >> 4;   // k-half select
    const int r    = lane & 15;   // row (A) / col (B,C,D) within tile
    const int mt   = wave >> 2;   // 0..1 : which 16-row tile
    const int nq   = wave & 3;    // 0..3 : N quarter
    const size_t rowBase = (size_t)blockIdx.x * ROWS;

    // ---- Phase 0: stage x tile into LDS as bf16 (rows are contiguous) ----
    for (int i = tid * 4; i < ROWS * DIMD; i += 256 * 4) {
        float4 v = *(const float4*)(x + rowBase * DIMD + i);
        xb[i + 0] = f2bf(v.x);
        xb[i + 1] = f2bf(v.y);
        xb[i + 2] = f2bf(v.z);
        xb[i + 3] = f2bf(v.w);
    }
    __syncthreads();

    // ---- Phase 1: Q = x @ Wq + bq -> qb (bf16). K=1024, wave does 256 cols.
    for (int nc = 0; nc < 4; ++nc) {
        const int ncol = nq * 256 + nc * 64;
        v8f acc[4] = {zero8(), zero8(), zero8(), zero8()};
        for (int kb = 0; kb < DIMD; kb += 32) {
            v16bf a = load_frag_a(xb + (mt * 16 + r) * DIMD + kb + 16 * h);
            __builtin_prefetch(WqT + (size_t)(ncol + r) * DIMD + kb + 256, 0, 1);
#pragma unroll
            for (int t = 0; t < 4; ++t) {
                v16bf b = load_frag_b(WqT + (size_t)(ncol + t * 16 + r) * DIMD + kb + 16 * h);
                acc[t] = wmma_bf16(a, b, acc[t]);
            }
        }
#pragma unroll
        for (int t = 0; t < 4; ++t) {
            const int col = ncol + t * 16 + r;
            const float bias = bq[col];
#pragma unroll
            for (int g = 0; g < 8; ++g) {
                const int m = mt * 16 + g + 8 * h;  // C/D row = vgpr + 8*half
                qb[m * DIMD + col] = f2bf(acc[t][g] + bias);
            }
        }
    }
    __syncthreads();

    // ---- Phase 2: per-row scale = 1/(max(||Q||,eps)*sqrt(d)) --------------
    if (tid < ROWS) {
        float ss = 0.f;
        const uint16_t* qr = qb + tid * DIMD;
        for (int c = 0; c < DIMD; ++c) {
            float v = bf2f(qr[c]);
            ss = fmaf(v, v, ss);
        }
        float nrm = fmaxf(sqrtf(ss), 1e-12f);
        qscale[tid] = (1.0f / nrm) * 0.03125f;  // 1/sqrt(1024)
    }
    __syncthreads();

    // ---- Phase 3: scores = Qn @ KnT -> wf (f32). N=512 slots, wave=128 ----
    for (int sc = 0; sc < 2; ++sc) {
        const int slot0 = nq * 128 + sc * 64;
        v8f acc[4] = {zero8(), zero8(), zero8(), zero8()};
        for (int kb = 0; kb < DIMD; kb += 32) {
            v16bf a = load_frag_a(qb + (mt * 16 + r) * DIMD + kb + 16 * h);
#pragma unroll
            for (int t = 0; t < 4; ++t) {
                v16bf b = load_frag_b(Kn + (size_t)(slot0 + t * 16 + r) * DIMD + kb + 16 * h);
                acc[t] = wmma_bf16(a, b, acc[t]);
            }
        }
#pragma unroll
        for (int t = 0; t < 4; ++t) {
            const int slot = slot0 + t * 16 + r;
#pragma unroll
            for (int g = 0; g < 8; ++g) {
                const int m = mt * 16 + g + 8 * h;
                wf[m * NSLOT + slot] = acc[t][g] * qscale[m];
            }
        }
    }
    __syncthreads();

    // ---- Phase 4: softmax rows of wf -> wb (bf16 probabilities) -----------
    if (tid < ROWS) {
        const float* row = wf + tid * NSLOT;
        float mx = -1e30f;
        for (int s = 0; s < NSLOT; ++s) mx = fmaxf(mx, row[s]);
        float sum = 0.f;
        for (int s = 0; s < NSLOT; ++s) sum += __expf(row[s] - mx);
        rmax[tid] = mx;
        rsum[tid] = 1.0f / sum;
    }
    __syncthreads();
    for (int i = tid; i < ROWS * NSLOT; i += 256) {
        const int m = i >> 9;  // /512
        wb[i] = f2bf(__expf(wf[i] - rmax[m]) * rsum[m]);
    }
    __syncthreads();

    // ---- Phase 5: R = w @ V -> qb (bf16, overwrites Q). K=512 -------------
    for (int dc = 0; dc < 4; ++dc) {
        const int col0 = nq * 256 + dc * 64;
        v8f acc[4] = {zero8(), zero8(), zero8(), zero8()};
        for (int kb = 0; kb < NSLOT; kb += 32) {
            v16bf a = load_frag_a(wb + (mt * 16 + r) * NSLOT + kb + 16 * h);
#pragma unroll
            for (int t = 0; t < 4; ++t) {
                v16bf b = load_frag_b(Vt + (size_t)(col0 + t * 16 + r) * NSLOT + kb + 16 * h);
                acc[t] = wmma_bf16(a, b, acc[t]);
            }
        }
#pragma unroll
        for (int t = 0; t < 4; ++t) {
            const int col = col0 + t * 16 + r;
#pragma unroll
            for (int g = 0; g < 8; ++g) {
                const int m = mt * 16 + g + 8 * h;
                qb[m * DIMD + col] = f2bf(acc[t][g]);
            }
        }
    }
    __syncthreads();

    // ---- Phase 6: g = sigmoid([x,R] @ Wg + bg); out = g*x + (1-g)*R -------
    for (int gc = 0; gc < 4; ++gc) {
        const int col0 = nq * 256 + gc * 64;
        v8f acc[4] = {zero8(), zero8(), zero8(), zero8()};
        // x half of the concat (K = 0..1023)
        for (int kb = 0; kb < DIMD; kb += 32) {
            v16bf a = load_frag_a(xb + (mt * 16 + r) * DIMD + kb + 16 * h);
            __builtin_prefetch(WgT + (size_t)(col0 + r) * 2048 + kb + 256, 0, 1);
#pragma unroll
            for (int t = 0; t < 4; ++t) {
                v16bf b = load_frag_b(WgT + (size_t)(col0 + t * 16 + r) * 2048 + kb + 16 * h);
                acc[t] = wmma_bf16(a, b, acc[t]);
            }
        }
        // R half of the concat (K = 1024..2047)
        for (int kb = 0; kb < DIMD; kb += 32) {
            v16bf a = load_frag_a(qb + (mt * 16 + r) * DIMD + kb + 16 * h);
#pragma unroll
            for (int t = 0; t < 4; ++t) {
                v16bf b = load_frag_b(WgT + (size_t)(col0 + t * 16 + r) * 2048 + DIMD + kb + 16 * h);
                acc[t] = wmma_bf16(a, b, acc[t]);
            }
        }
#pragma unroll
        for (int t = 0; t < 4; ++t) {
            const int col = col0 + t * 16 + r;
            const float bias = bg[col];
#pragma unroll
            for (int g = 0; g < 8; ++g) {
                const int m = mt * 16 + g + 8 * h;
                const float gate = 1.0f / (1.0f + __expf(-(acc[t][g] + bias)));
                const float xv = x[(rowBase + m) * DIMD + col];
                const float rv = bf2f(qb[m * DIMD + col]);
                out[(rowBase + m) * DIMD + col] = gate * xv + (1.0f - gate) * rv;
            }
        }
    }
}

// ---------------------------------------------------------------------------
extern "C" void kernel_launch(void* const* d_in, const int* in_sizes, int n_in,
                              void* d_out, int out_size, void* d_ws, size_t ws_size,
                              hipStream_t stream) {
    (void)in_sizes; (void)n_in; (void)out_size; (void)ws_size;
    const float* x    = (const float*)d_in[0];  // [8,4096,1024]
    const float* memv = (const float*)d_in[1];  // [512,1024]
    const float* Wq   = (const float*)d_in[2];
    const float* bq   = (const float*)d_in[3];
    const float* Wk   = (const float*)d_in[4];
    const float* bk   = (const float*)d_in[5];
    const float* Wv   = (const float*)d_in[6];
    const float* bv   = (const float*)d_in[7];
    const float* Wg   = (const float*)d_in[8];  // [2048,1024]
    const float* bg   = (const float*)d_in[9];
    float* out = (float*)d_out;

    // Workspace layout (bf16), ~8 MB total
    uint16_t* WqT = (uint16_t*)d_ws;                 // [1024][1024]
    uint16_t* WgT = WqT + (size_t)DIMD * DIMD;       // [1024][2048]
    uint16_t* Kn  = WgT + (size_t)DIMD * 2 * DIMD;   // [512][1024]
    uint16_t* Vt  = Kn + (size_t)NSLOT * DIMD;       // [1024][512]

    convert_wT_kernel<<<(DIMD * DIMD + 255) / 256, 256, 0, stream>>>(Wq, WqT, DIMD, DIMD);
    convert_wT_kernel<<<(2 * DIMD * DIMD + 255) / 256, 256, 0, stream>>>(Wg, WgT, 2 * DIMD, DIMD);
    kv_kernel<<<NSLOT, 256, 0, stream>>>(memv, Wk, bk, Wv, bv, Kn, Vt);

    const size_t ldsBytes =
        (size_t)ROWS * DIMD * 2 +   // xb
        (size_t)ROWS * DIMD * 2 +   // qb
        (size_t)ROWS * NSLOT * 4 +  // wf
        (size_t)ROWS * NSLOT * 2 +  // wb
        (size_t)ROWS * 4 * 3;       // qscale, rmax, rsum
    const int nTokens = 8 * 4096;
    fused_kernel<<<nTokens / ROWS, 256, ldsBytes, stream>>>(
        x, bq, bg, WqT, WgT, Kn, Vt, out);
}
